// LinearTemporalCrossAttention_21715354649987
// MI455X (gfx1250) — compile-verified, matrix operand order checked by reading, not run on hardware
//
#include <hip/hip_runtime.h>

// LinearTemporalCrossAttention for MI455X (gfx1250, wave32, WMMA + TDM).
// Pipeline: LN -> (bf16) -> 3x GEMM (v_wmma_f32_16x16x32_bf16, TDM-fed LDS,
// double buffered) -> k seq-softmax stats -> att = K_sm^T V (WMMA) ->
// y = softmax(q) att (WMMA).

typedef unsigned short u16;
typedef __attribute__((ext_vector_type(16))) __bf16 v16bf;
typedef __attribute__((ext_vector_type(8)))  __bf16 v8bf;
typedef __attribute__((ext_vector_type(8)))  float  v8f;
typedef __attribute__((ext_vector_type(4)))  unsigned int u32x4;
typedef __attribute__((ext_vector_type(4)))  int i32x4;
typedef __attribute__((ext_vector_type(8)))  int i32x8;

#define T_SEQ 2048
#define B_BATCH 8
#define D_MODEL 1024
#define N_HEAD 16
#define D_HEAD 64
#define M_TOK (T_SEQ * B_BATCH)   // 16384
#define LDSP 40                    // padded LDS row stride (bf16), 80B = 5x16B

// Tensor Data Mover path: guarded so absence of the builtin falls back to the
// proven global->VGPR->ds_store staging. Device pass only.
#if defined(__AMDGCN__) && defined(__has_builtin)
#if __has_builtin(__builtin_amdgcn_tensor_load_to_lds) && \
    __has_builtin(__builtin_amdgcn_s_wait_tensorcnt)
#define USE_TDM 1
#endif
#endif
#ifndef USE_TDM
#define USE_TDM 0
#endif

static __device__ __forceinline__ u16 f32_to_bf16(float f) {
  unsigned int u = __float_as_uint(f);
  u += 0x7FFFu + ((u >> 16) & 1u);           // round-to-nearest-even
  return (u16)(u >> 16);
}

// Load one 16x32 bf16 A/B fragment row-slice for this lane.
// p points at the lane's row base within the 32-wide K window; kb=(lane>>4)*8.
static __device__ __forceinline__ v16bf frag_ld(const u16* p, int kb) {
  uint4 lo = *(const uint4*)(p + kb);
  uint4 hi = *(const uint4*)(p + kb + 16);
  v8bf l8 = __builtin_bit_cast(v8bf, lo);
  v8bf h8 = __builtin_bit_cast(v8bf, hi);
  v16bf f;
#pragma unroll
  for (int i = 0; i < 8; ++i) { f[i] = l8[i]; f[i + 8] = h8[i]; }
  return f;
}

static __device__ __forceinline__ v8f wmma_bf16(v16bf a, v16bf b, v8f c) {
  return __builtin_amdgcn_wmma_f32_16x16x32_bf16(false, a, false, b, (short)0, c,
                                                 false, false);
}

#if USE_TDM
// Issue one TDM 2D tile load: tile_rows x 32 bf16 elems, row stride D_MODEL,
// into LDS with padding 16DW+4DW => padded row stride of LDSP(=40) halves.
// D# layout per CDNA5 ISA ch.8 (group0: count/lds/global/type, group1: dims).
static __device__ __forceinline__ void tdm_load_tile_2d(const u16* lds_dst,
                                                        const u16* gsrc,
                                                        int tile_rows,
                                                        unsigned tensor_rows) {
  const unsigned long long ga = (unsigned long long)(size_t)gsrc;
  u32x4 g0;
  g0.x = 1u;                                          // count=1, user mode
  g0.y = (unsigned)(size_t)(const void*)lds_dst;      // LDS byte address
  g0.z = (unsigned)ga;                                // global addr [31:0]
  g0.w = (unsigned)((ga >> 32) & 0x1ffffffu) | (2u << 30);  // [56:32] | type=2
  i32x8 g1;
  // data_size=2B(code1), pad_enable, pad_interval=16DW(code3), pad_amount=4DW(code3)
  g1[0] = (int)((1u << 16) | (1u << 20) | (3u << 22) | (3u << 25));
  g1[1] = (int)(((unsigned)D_MODEL & 0xffffu) << 16);       // tensor_dim0 lo16
  g1[2] = (int)((tensor_rows & 0xffffu) << 16);             // dim0 hi=0 | tensor_dim1 lo16
  g1[3] = (int)(((tensor_rows >> 16) & 0xffffu) | (32u << 16)); // dim1 hi | tile_dim0=32
  g1[4] = (int)(tile_rows & 0xffff);                        // tile_dim1 | tile_dim2=0
  g1[5] = (int)D_MODEL;                                     // tensor_dim0_stride lo32
  g1[6] = 0;                                                // stride hi | dim1_stride lo
  g1[7] = 0;
  const i32x4 z4 = {0, 0, 0, 0};
#if __clang_major__ >= 23
  const i32x8 z8 = {0, 0, 0, 0, 0, 0, 0, 0};
  __builtin_amdgcn_tensor_load_to_lds(g0, g1, z4, z4, z8, 0);
#else
  __builtin_amdgcn_tensor_load_to_lds(g0, g1, z4, z4, 0);
#endif
}
#endif

// ---------------------------------------------------------------- LayerNorm
__global__ __launch_bounds__(256) void ln_kernel(
    const float* __restrict__ x, const float* __restrict__ g,
    const float* __restrict__ bB, const float* __restrict__ tg,
    const float* __restrict__ tb, u16* __restrict__ xn, u16* __restrict__ xtn) {
  const int r = blockIdx.x * 8 + (threadIdx.x >> 5);   // row = b*T + t
  const int lane = threadIdx.x & 31;
  const int b = r >> 11, tt = r & (T_SEQ - 1);
  const float* row = x + ((size_t)tt * B_BATCH + b) * D_MODEL;
  float v[32];
  float s = 0.f, sq = 0.f;
#pragma unroll
  for (int i = 0; i < 32; ++i) {
    v[i] = row[lane + i * 32];
    s += v[i]; sq += v[i] * v[i];
  }
#pragma unroll
  for (int m = 16; m >= 1; m >>= 1) {
    s  += __shfl_xor(s, m, 32);
    sq += __shfl_xor(sq, m, 32);
  }
  const float mean = s * (1.f / D_MODEL);
  const float var  = sq * (1.f / D_MODEL) - mean * mean;
  const float rstd = rsqrtf(var + 1e-5f);
  const size_t ob = (size_t)r * D_MODEL;
#pragma unroll
  for (int i = 0; i < 32; ++i) {
    const int d = lane + i * 32;
    const float nv = (v[i] - mean) * rstd;
    xn[ob + d]  = f32_to_bf16(nv * g[d]  + bB[d]);
    xtn[ob + d] = f32_to_bf16(nv * tg[d] + tb[d]);
  }
}

// ------------------------------------------------------- weight f32 -> bf16
__global__ __launch_bounds__(256) void cvt_bf16(const float* __restrict__ src,
                                                u16* __restrict__ dst, int n) {
  const int i = blockIdx.x * 256 + threadIdx.x;
  if (i < n) dst[i] = f32_to_bf16(src[i]);
}

// -------------------------------------------------------------------- GEMM
// C[m,i] = sum_j A[m,j]*W[i,j] + bias[i].  Block tile 128(M) x 128(N),
// 8 waves as 4x2, 32x64 per wave (2x4 WMMA tiles), K-step 32, double-buffered
// LDS fed by the Tensor Data Mover (fallback: manual staging).
__global__ __launch_bounds__(256) void gemm_bf16(
    const u16* __restrict__ A, const u16* __restrict__ W,
    const float* __restrict__ bias, float* __restrict__ C) {
  __shared__ __attribute__((aligned(16))) u16 lsA[2][128 * LDSP];
  __shared__ __attribute__((aligned(16))) u16 lsB[2][128 * LDSP];
  const int t = threadIdx.x;
  const int lane = t & 31, wave = t >> 5;
  const int wm = wave & 3, wn = wave >> 2;      // 4(M) x 2(N)
  const int gm0 = blockIdx.x * 128;
  const int gn0 = blockIdx.y * 128;
  const int kb = (lane >> 4) * 8;
  const int fm = wm * 32 + (lane & 15);
  const int fn = wn * 64 + (lane & 15);
  v8f acc[2][4] = {};

#if USE_TDM
  if (wave == 0) {
    tdm_load_tile_2d(lsA[0], A + (size_t)gm0 * D_MODEL, 128, M_TOK);
    tdm_load_tile_2d(lsB[0], W + (size_t)gn0 * D_MODEL, 128, D_MODEL);
  }
  for (int k0 = 0; k0 < D_MODEL; k0 += 32) {
    const int cur = (k0 >> 5) & 1;
    __syncthreads();                 // all reads of buf[1-cur] finished
    if (wave == 0) {
      if (k0 + 32 < D_MODEL) {
        tdm_load_tile_2d(lsA[1 - cur],
                         A + (size_t)gm0 * D_MODEL + (k0 + 32), 128, M_TOK);
        tdm_load_tile_2d(lsB[1 - cur],
                         W + (size_t)gn0 * D_MODEL + (k0 + 32), 128, D_MODEL);
        __builtin_amdgcn_s_wait_tensorcnt(2);   // pair for buf[cur] done
      } else {
        __builtin_amdgcn_s_wait_tensorcnt(0);
      }
    }
    __syncthreads();                 // publish buf[cur]
    const u16* bA = lsA[cur];
    const u16* bB = lsB[cur];
    v16bf af0 = frag_ld(bA + (size_t)fm * LDSP, kb);
    v16bf af1 = frag_ld(bA + (size_t)(fm + 16) * LDSP, kb);
    v16bf bfj[4];
#pragma unroll
    for (int j = 0; j < 4; ++j)
      bfj[j] = frag_ld(bB + (size_t)(fn + j * 16) * LDSP, kb);
#pragma unroll
    for (int j = 0; j < 4; ++j) {
      acc[0][j] = wmma_bf16(af0, bfj[j], acc[0][j]);
      acc[1][j] = wmma_bf16(af1, bfj[j], acc[1][j]);
    }
  }
#else
  const int srow = t >> 2;                 // 0..63
  const int schk = (t & 3) * 8;            // 0/8/16/24
  for (int k0 = 0; k0 < D_MODEL; k0 += 32) {
    const uint4 a0 = *(const uint4*)(A + (size_t)(gm0 + srow) * D_MODEL + k0 + schk);
    const uint4 a1 = *(const uint4*)(A + (size_t)(gm0 + srow + 64) * D_MODEL + k0 + schk);
    const uint4 b0 = *(const uint4*)(W + (size_t)(gn0 + srow) * D_MODEL + k0 + schk);
    const uint4 b1 = *(const uint4*)(W + (size_t)(gn0 + srow + 64) * D_MODEL + k0 + schk);
    if (k0 + 32 < D_MODEL) {               // gfx1250: global_prefetch_b8
      __builtin_prefetch(A + (size_t)(gm0 + srow) * D_MODEL + k0 + 32 + schk, 0, 3);
      __builtin_prefetch(W + (size_t)(gn0 + srow) * D_MODEL + k0 + 32 + schk, 0, 3);
    }
    __syncthreads();
    *(uint4*)(lsA[0] + srow * LDSP + schk) = a0;
    *(uint4*)(lsA[0] + (srow + 64) * LDSP + schk) = a1;
    *(uint4*)(lsB[0] + srow * LDSP + schk) = b0;
    *(uint4*)(lsB[0] + (srow + 64) * LDSP + schk) = b1;
    __syncthreads();
    v16bf af0 = frag_ld(lsA[0] + (size_t)fm * LDSP, kb);
    v16bf af1 = frag_ld(lsA[0] + (size_t)(fm + 16) * LDSP, kb);
    v16bf bfj[4];
#pragma unroll
    for (int j = 0; j < 4; ++j)
      bfj[j] = frag_ld(lsB[0] + (size_t)(fn + j * 16) * LDSP, kb);
#pragma unroll
    for (int j = 0; j < 4; ++j) {
      acc[0][j] = wmma_bf16(af0, bfj[j], acc[0][j]);
      acc[1][j] = wmma_bf16(af1, bfj[j], acc[1][j]);
    }
  }
#endif
#pragma unroll
  for (int i = 0; i < 2; ++i)
#pragma unroll
    for (int j = 0; j < 4; ++j) {
      const int n = gn0 + wn * 64 + j * 16 + (lane & 15);
      const int mb = gm0 + wm * 32 + i * 16 + (lane >> 4) * 8;
      const float bn = bias[n];
#pragma unroll
      for (int r = 0; r < 8; ++r)
        C[(size_t)(mb + r) * D_MODEL + n] = acc[i][j][r] + bn;
    }
}

// ------------------------------------- per-(b,h,d) online max / sum-exp over T
__global__ __launch_bounds__(256) void kstats_kernel(
    const float* __restrict__ kf, float* __restrict__ kmax,
    float* __restrict__ ksum) {
  const int bh = blockIdx.x;               // b*16 + h
  const int b = bh >> 4, h = bh & 15;
  const int d = threadIdx.x & 63;
  const int seg = threadIdx.x >> 6;        // 4 segments of 512 rows
  const float* base = kf + ((size_t)b * T_SEQ) * D_MODEL + h * D_HEAD + d;
  float m = -1e30f, s = 0.f;
  for (int n = seg * 512; n < seg * 512 + 512; ++n) {
    const float v = base[(size_t)n * D_MODEL];
    const float mn = fmaxf(m, v);
    s = s * __expf(m - mn) + __expf(v - mn);
    m = mn;
  }
  __shared__ float sm[4][64], ss[4][64];
  sm[seg][d] = m; ss[seg][d] = s;
  __syncthreads();
  if (seg == 0) {
#pragma unroll
    for (int i = 1; i < 4; ++i) {
      const float m2 = sm[i][d], s2 = ss[i][d];
      const float mn = fmaxf(m, m2);
      s = s * __expf(m - mn) + s2 * __expf(m2 - mn);
      m = mn;
    }
    kmax[bh * 64 + d] = m;
    ksum[bh * 64 + d] = s;
  }
}

// ------------------------- att[b,h] = K_sm^T V  (M=64 d, N=64 l, K=T=2048)
__global__ __launch_bounds__(256) void att_kernel(
    const float* __restrict__ kf, const float* __restrict__ vf,
    const float* __restrict__ kmax, const float* __restrict__ ksum,
    u16* __restrict__ attb) {
  const int bh = blockIdx.x;
  const int b = bh >> 4, h = bh & 15;
  __shared__ __attribute__((aligned(16))) u16 lsK[64 * LDSP];  // [d][n]
  __shared__ __attribute__((aligned(16))) u16 lsV[64 * LDSP];  // [l][n]
  const int t = threadIdx.x, lane = t & 31, wave = t >> 5;
  const int wm = wave & 3, wn = wave >> 2; // wave tile: 16(d) x 32(l)
  const int d = t & 63, nl = t >> 6;
  const float km = kmax[bh * 64 + d];
  const float inv = 1.0f / ksum[bh * 64 + d];
  const size_t rowb = ((size_t)b * T_SEQ) * D_MODEL + h * D_HEAD;
  const int kb = (lane >> 4) * 8;
  const int am = wm * 16 + (lane & 15);
  v8f acc[2] = {};
  for (int n0 = 0; n0 < T_SEQ; n0 += 32) {
    __syncthreads();
#pragma unroll
    for (int p = 0; p < 8; ++p) {
      const int n = nl + p * 4;
      const float kv = kf[rowb + (size_t)(n0 + n) * D_MODEL + d];
      lsK[d * LDSP + n] = f32_to_bf16(__expf(kv - km) * inv);
      lsV[d * LDSP + n] = f32_to_bf16(vf[rowb + (size_t)(n0 + n) * D_MODEL + d]);
    }
    __syncthreads();
    v16bf af  = frag_ld(lsK + (size_t)am * LDSP, kb);
    v16bf bv0 = frag_ld(lsV + (size_t)(wn * 32 + (lane & 15)) * LDSP, kb);
    v16bf bv1 = frag_ld(lsV + (size_t)(wn * 32 + 16 + (lane & 15)) * LDSP, kb);
    acc[0] = wmma_bf16(af, bv0, acc[0]);
    acc[1] = wmma_bf16(af, bv1, acc[1]);
  }
#pragma unroll
  for (int j = 0; j < 2; ++j) {
    const int col = wn * 32 + j * 16 + (lane & 15);
    const int rb = wm * 16 + (lane >> 4) * 8;
#pragma unroll
    for (int r = 0; r < 8; ++r)
      attb[((size_t)bh * 64 + rb + r) * 64 + col] = f32_to_bf16(acc[j][r]);
  }
}

// ------------------------- y = softmax_d(q) @ att  (per (b,h), M-tile of 64)
__global__ __launch_bounds__(256) void y_kernel(
    const float* __restrict__ qf, const u16* __restrict__ attb,
    float* __restrict__ out) {
  const int blk = blockIdx.x;
  const int mt = blk & 31;                 // 32 tiles of 64 rows over T
  const int bh = blk >> 5;
  const int b = bh >> 4, h = bh & 15;
  __shared__ float lq[64 * 64];            // q tile, f32
  __shared__ __attribute__((aligned(16))) u16 lqb[64 * 72];  // softmax(q) bf16
  __shared__ __attribute__((aligned(16))) u16 latt[64 * 72]; // att^T bf16 [l][d]
  const int t = threadIdx.x, lane = t & 31, wave = t >> 5;
  const size_t qbase = ((size_t)b * T_SEQ + mt * 64) * D_MODEL + h * D_HEAD;
  for (int idx = t; idx < 4096; idx += 256) {
    const int r = idx >> 6, c = idx & 63;
    lq[r * 64 + c] = qf[qbase + (size_t)r * D_MODEL + c];
    latt[c * 72 + r] = attb[((size_t)bh * 64 + r) * 64 + c];  // transpose
  }
  __syncthreads();
  if (t < 64) {                            // row softmax over d=64
    const int r = t;
    float m = -1e30f;
    for (int c = 0; c < 64; ++c) m = fmaxf(m, lq[r * 64 + c]);
    float s = 0.f;
    for (int c = 0; c < 64; ++c) s += __expf(lq[r * 64 + c] - m);
    const float is = 1.0f / s;
    for (int c = 0; c < 64; ++c)
      lqb[r * 72 + c] = f32_to_bf16(__expf(lq[r * 64 + c] - m) * is);
  }
  __syncthreads();
  const int wm = wave & 3, wn = wave >> 2; // wave tile 16(m) x 32(l)
  const int kb = (lane >> 4) * 8;
  v8f acc[2] = {};
#pragma unroll
  for (int k0 = 0; k0 < 64; k0 += 32) {
    v16bf af  = frag_ld(lqb + (size_t)(wm * 16 + (lane & 15)) * 72 + k0, kb);
    v16bf bt0 = frag_ld(latt + (size_t)(wn * 32 + (lane & 15)) * 72 + k0, kb);
    v16bf bt1 = frag_ld(latt + (size_t)(wn * 32 + 16 + (lane & 15)) * 72 + k0, kb);
    acc[0] = wmma_bf16(af, bt0, acc[0]);
    acc[1] = wmma_bf16(af, bt1, acc[1]);
  }
#pragma unroll
  for (int j = 0; j < 2; ++j) {
    const int l = wn * 32 + j * 16 + (lane & 15);
    const int rb = wm * 16 + (lane >> 4) * 8;
#pragma unroll
    for (int r = 0; r < 8; ++r)
      out[((size_t)b * T_SEQ + mt * 64 + rb + r) * D_MODEL + h * D_HEAD + l] =
          acc[j][r];
  }
}

extern "C" void kernel_launch(void* const* d_in, const int* in_sizes, int n_in,
                              void* d_out, int out_size, void* d_ws,
                              size_t ws_size, hipStream_t stream) {
  (void)in_sizes; (void)n_in; (void)out_size; (void)ws_size;
  const float* x     = (const float*)d_in[0];
  const float* ln_g  = (const float*)d_in[1];
  const float* ln_b  = (const float*)d_in[2];
  const float* tln_g = (const float*)d_in[3];
  const float* tln_b = (const float*)d_in[4];
  const float* Wq = (const float*)d_in[5];
  const float* bq = (const float*)d_in[6];
  const float* Wk = (const float*)d_in[7];
  const float* bk = (const float*)d_in[8];
  const float* Wv = (const float*)d_in[9];
  const float* bv = (const float*)d_in[10];
  float* out = (float*)d_out;

  char* ws = (char*)d_ws;
  size_t cur = 0;
  auto alloc = [&](size_t bytes) -> void* {
    void* p = (void*)(ws + cur);
    cur += (bytes + 255) & ~(size_t)255;
    return p;
  };
  u16* xn   = (u16*)alloc((size_t)M_TOK * D_MODEL * 2);
  u16* xtn  = (u16*)alloc((size_t)M_TOK * D_MODEL * 2);
  u16* wqb  = (u16*)alloc((size_t)D_MODEL * D_MODEL * 2);
  u16* wkb  = (u16*)alloc((size_t)D_MODEL * D_MODEL * 2);
  u16* wvb  = (u16*)alloc((size_t)D_MODEL * D_MODEL * 2);
  float* qf = (float*)alloc((size_t)M_TOK * D_MODEL * 4);
  float* kf = (float*)alloc((size_t)M_TOK * D_MODEL * 4);
  float* vf = (float*)alloc((size_t)M_TOK * D_MODEL * 4);
  float* kmax = (float*)alloc((size_t)B_BATCH * N_HEAD * D_HEAD * 4);
  float* ksum = (float*)alloc((size_t)B_BATCH * N_HEAD * D_HEAD * 4);
  u16* attb = (u16*)alloc((size_t)B_BATCH * N_HEAD * D_HEAD * D_HEAD * 2);

  ln_kernel<<<M_TOK / 8, 256, 0, stream>>>(x, ln_g, ln_b, tln_g, tln_b, xn, xtn);
  const int nw = D_MODEL * D_MODEL;
  cvt_bf16<<<(nw + 255) / 256, 256, 0, stream>>>(Wq, wqb, nw);
  cvt_bf16<<<(nw + 255) / 256, 256, 0, stream>>>(Wk, wkb, nw);
  cvt_bf16<<<(nw + 255) / 256, 256, 0, stream>>>(Wv, wvb, nw);
  dim3 ggrid(M_TOK / 128, D_MODEL / 128);
  gemm_bf16<<<ggrid, 256, 0, stream>>>(xn,  wqb, bq, qf);
  gemm_bf16<<<ggrid, 256, 0, stream>>>(xtn, wkb, bk, kf);
  gemm_bf16<<<ggrid, 256, 0, stream>>>(xtn, wvb, bv, vf);
  kstats_kernel<<<B_BATCH * N_HEAD, 256, 0, stream>>>(kf, kmax, ksum);
  att_kernel<<<B_BATCH * N_HEAD, 256, 0, stream>>>(kf, vf, kmax, ksum, attb);
  y_kernel<<<B_BATCH * N_HEAD * (T_SEQ / 64), 256, 0, stream>>>(qf, attb, out);
}
